// CrossAttFeatTrans_23845658427611
// MI455X (gfx1250) — matrix operand order repaired and checked by reading, not compiled, not run on hardware
//
#include <hip/hip_runtime.h>
#include <math.h>

// ---------------------------------------------------------------------------
// CDNA5 (gfx1250) implementation: bf16 WMMA everywhere, wave32, f32 accumulate
// ---------------------------------------------------------------------------

typedef __attribute__((ext_vector_type(16))) __bf16 v16bf;
typedef __attribute__((ext_vector_type(8)))  float  v8f;

#define BB 4
#define U1N 2048
#define U2N 2048
#define CC 256
#define FF 256
#define MM 4
#define DD 64

__device__ __forceinline__ v8f wmma_bf16(v16bf a, v16bf b, v8f c) {
  // D = A(16x32 bf16) x B(32x16 bf16) + C(16x16 f32)
  return __builtin_amdgcn_wmma_f32_16x16x32_bf16(false, a, false, b, (short)0, c,
                                                 false, false);
}

// A fragment (16x32, row-major source): lane lo = row; lanes 0-15 hold
// K = {klo..klo+7, 16+klo..}, klo = 8*(lane>=16). p -> &src[row][k0+klo].
__device__ __forceinline__ v16bf afrag_f32(const float* p) {
  v16bf x;
#pragma unroll
  for (int t = 0; t < 8; ++t) { x[t] = (__bf16)p[t]; x[8 + t] = (__bf16)p[16 + t]; }
  return x;
}
__device__ __forceinline__ v16bf afrag_bf16(const __bf16* p) {
  v16bf x;
#pragma unroll
  for (int t = 0; t < 8; ++t) { x[t] = p[t]; x[8 + t] = p[16 + t]; }
  return x;
}
// B fragment (32x16) from W stored [N][K] row-major: lane lo = n; lanes 0-15
// hold K = {klo..klo+7, 16+klo..}, klo = 8*(lane>=16). p -> &W[n][k0 + klo].
__device__ __forceinline__ v16bf bfrag_f32(const float* p) {
  v16bf x;
#pragma unroll
  for (int t = 0; t < 8; ++t) { x[t] = (__bf16)p[t]; x[8 + t] = (__bf16)p[16 + t]; }
  return x;
}
__device__ __forceinline__ v16bf bfrag_bf16(const __bf16* p) {
  v16bf x;
#pragma unroll
  for (int t = 0; t < 8; ++t) { x[t] = p[t]; x[8 + t] = p[16 + t]; }
  return x;
}

// ---------------------------------------------------------------------------
// Kernel 1: projection GEMM  out = bf16(X @ W^T + bias)
//   X: [nrows, 256] f32, W: [N, 256] f32 row-major, out bf16.
//   transpose_store=1 (V proj): out[(b*1024 + n)*2048 + j], row = b*2048 + j.
// Each wave computes one 16x16 output tile, 8 WMMA k-steps.
// ---------------------------------------------------------------------------
__global__ __launch_bounds__(128) void cafe_proj_gemm(
    const float* __restrict__ X, const float* __restrict__ W,
    const float* __restrict__ bias, __bf16* __restrict__ out,
    int nrows, int N, int transpose_store) {
  const int lane = threadIdx.x & 31;
  const int wid  = threadIdx.x >> 5;
  const int lo   = lane & 15;
  const int hi   = lane >> 4;
  const int klo  = hi * 8;
  const int kbB  = hi * 8;

  const int tile = blockIdx.x * 4 + wid;
  const int ntn  = N >> 4;
  const int row0 = (tile / ntn) * 16;
  const int n0   = (tile % ntn) * 16;

  const float* Arow = X + (size_t)(row0 + lo) * CC;
  const float* Brow = W + (size_t)(n0 + lo) * CC;

  v8f acc = {};
#pragma unroll
  for (int k0 = 0; k0 < CC; k0 += 32) {
    v16bf a = afrag_f32(Arow + k0 + klo);
    v16bf b = bfrag_f32(Brow + k0 + kbB);
    acc = wmma_bf16(a, b, acc);
  }
  const int n = n0 + lo;
  const float bv = bias ? bias[n] : 0.0f;
#pragma unroll
  for (int g = 0; g < 8; ++g) {
    const int row = row0 + g + 8 * hi;
    const float v = acc[g] + bv;
    size_t addr;
    if (transpose_store) {
      const int bb = row / U2N, j = row % U2N;
      addr = ((size_t)bb * (MM * FF) + n) * U2N + j;
    } else {
      addr = (size_t)row * N + n;
    }
    out[addr] = (__bf16)v;
  }
}

// ---------------------------------------------------------------------------
// Kernel 2: attention for one (b, m, 16-row i-tile) per workgroup (8 waves).
//   scores = clip((Q Kt)/8, +-500); softmax over j; fused = P V.
//   Flash-style: pass A = row max (scores discarded), then chunked
//   recompute -> exp into LDS bf16 tile -> WMMA against vt; final 1/rowsum.
// ---------------------------------------------------------------------------
__global__ __launch_bounds__(256) void cafe_attn(
    const __bf16* __restrict__ qb, const __bf16* __restrict__ kb,
    const __bf16* __restrict__ vt, float* __restrict__ fused) {
  __shared__ __bf16 s_p[16][256];   // prob chunk (16 rows x 256 j)  = 8 KB
  __shared__ float  s_red[16][8];   // cross-wave row reductions

  const int lane = threadIdx.x & 31;
  const int wid  = threadIdx.x >> 5;
  const int lo   = lane & 15;
  const int hi   = lane >> 4;
  const int klo  = hi * 8;
  const int kbB  = hi * 8;

  const int it = blockIdx.x & 127;         // U1/16 = 128 tiles
  const int bm = blockIdx.x >> 7;          // b*M + m
  const int b  = bm >> 2;
  const int m  = bm & 3;
  const int i0 = it * 16;

  // Q A-fragments for the two D k-steps (D = 64)
  const __bf16* qrow = qb + (size_t)(b * U1N + i0 + lo) * CC + m * DD;
  const v16bf qa0 = afrag_bf16(qrow + 0  + klo);
  const v16bf qa1 = afrag_bf16(qrow + 32 + klo);
  const __bf16* kbase = kb + (size_t)(b * U2N) * CC + m * DD;

  // -------- pass A: running row max over all 128 j-tiles (split by wave) ----
  float rmax[8];
#pragma unroll
  for (int g = 0; g < 8; ++g) rmax[g] = -3.0e38f;
  for (int jt = wid; jt < 128; jt += 8) {
    const __bf16* krow = kbase + (size_t)(jt * 16 + lo) * CC;
    v8f sc = {};
    sc = wmma_bf16(qa0, bfrag_bf16(krow + 0  + kbB), sc);
    sc = wmma_bf16(qa1, bfrag_bf16(krow + 32 + kbB), sc);
#pragma unroll
    for (int g = 0; g < 8; ++g) {
      float s = fminf(fmaxf(sc[g] * 0.125f, -500.0f), 500.0f);
      s = fmaxf(s, __shfl_xor(s, 1, 32));
      s = fmaxf(s, __shfl_xor(s, 2, 32));
      s = fmaxf(s, __shfl_xor(s, 4, 32));
      s = fmaxf(s, __shfl_xor(s, 8, 32));
      rmax[g] = fmaxf(rmax[g], s);
    }
  }
  if (lo == 0) {
#pragma unroll
    for (int g = 0; g < 8; ++g) s_red[g + 8 * hi][wid] = rmax[g];
  }
  __syncthreads();
#pragma unroll
  for (int g = 0; g < 8; ++g) {
    float v = -3.0e38f;
#pragma unroll
    for (int w = 0; w < 8; ++w) v = fmaxf(v, s_red[g + 8 * hi][w]);
    rmax[g] = v;
  }
  __syncthreads();  // s_red reused for row sums below

  // -------- chunked exp + P*V -------------------------------------------
  float rs[8];
#pragma unroll
  for (int g = 0; g < 8; ++g) rs[g] = 0.0f;
  v8f acc0 = {}, acc1 = {};
  const int f0a = (wid * 2 + 0) * 16;
  const int f0b = (wid * 2 + 1) * 16;
  const __bf16* vbase = vt + (size_t)bm * FF * U2N;  // rows f, ld = U2

  for (int c = 0; c < 8; ++c) {            // 8 chunks of 256 j each
#pragma unroll
    for (int t = 0; t < 2; ++t) {          // 2 j-tiles per wave per chunk
      const int jt = c * 16 + wid * 2 + t;
      const __bf16* krow = kbase + (size_t)(jt * 16 + lo) * CC;
      v8f sc = {};
      sc = wmma_bf16(qa0, bfrag_bf16(krow + 0  + kbB), sc);
      sc = wmma_bf16(qa1, bfrag_bf16(krow + 32 + kbB), sc);
#pragma unroll
      for (int g = 0; g < 8; ++g) {
        const float s = fminf(fmaxf(sc[g] * 0.125f, -500.0f), 500.0f);
        const float e = __expf(s - rmax[g]);
        s_p[g + 8 * hi][(jt & 15) * 16 + lo] = (__bf16)e;
        float es = e;
        es += __shfl_xor(es, 1, 32);
        es += __shfl_xor(es, 2, 32);
        es += __shfl_xor(es, 4, 32);
        es += __shfl_xor(es, 8, 32);
        rs[g] += es;
      }
    }
    __syncthreads();
    const int kg0 = c * 256;
#pragma unroll
    for (int k0 = 0; k0 < 256; k0 += 32) {
      const v16bf pa = afrag_bf16(&s_p[lo][k0 + klo]);
      const __bf16* vp0 = vbase + (size_t)(f0a + lo) * U2N + kg0 + k0 + kbB;
      const __bf16* vp1 = vbase + (size_t)(f0b + lo) * U2N + kg0 + k0 + kbB;
      __builtin_prefetch(vp0 + 256, 0, 0);   // global_prefetch_b8
      acc0 = wmma_bf16(pa, bfrag_bf16(vp0), acc0);
      acc1 = wmma_bf16(pa, bfrag_bf16(vp1), acc1);
    }
    __syncthreads();
  }

  // -------- finalize: divide by row sum, store fused [b,m,i,f] f32 --------
  if (lo == 0) {
#pragma unroll
    for (int g = 0; g < 8; ++g) s_red[g + 8 * hi][wid] = rs[g];
  }
  __syncthreads();
  float rinv[8];
#pragma unroll
  for (int g = 0; g < 8; ++g) {
    float v = 0.0f;
#pragma unroll
    for (int w = 0; w < 8; ++w) v += s_red[g + 8 * hi][w];
    rinv[g] = 1.0f / v;
  }
  float* fbase = fused + ((size_t)bm * U1N + i0) * FF;
#pragma unroll
  for (int g = 0; g < 8; ++g) {
    const int r = g + 8 * hi;
    fbase[(size_t)r * FF + f0a + lo] = acc0[g] * rinv[g];
    fbase[(size_t)r * FF + f0b + lo] = acc1[g] * rinv[g];
  }
}

// ---------------------------------------------------------------------------
// Kernel 3: mid = gelu(fused @ W_mid^T + b_mid); out = mid @ W_out[m]^T + b_out;
//           layernorm(out) -> outn.  One 16-row block per workgroup (8 waves).
// ---------------------------------------------------------------------------
__global__ __launch_bounds__(256) void cafe_mid_out(
    const float* __restrict__ fused, const float* __restrict__ W_mid,
    const float* __restrict__ b_mid, const float* __restrict__ W_out,
    const float* __restrict__ b_out, const float* __restrict__ ln_g,
    const float* __restrict__ ln_b, float* __restrict__ outn) {
  __shared__ __bf16 s_mid[16][FF];   // 8 KB
  __shared__ float  s_out[16][FF];   // 16 KB
  __shared__ float  s_r1[16][16];
  __shared__ float  s_r2[16][16];

  const int lane = threadIdx.x & 31;
  const int wid  = threadIdx.x >> 5;
  const int lo   = lane & 15;
  const int hi   = lane >> 4;
  const int klo  = hi * 8;
  const int kbB  = hi * 8;

  const int r0 = blockIdx.x * 16;           // row in flat [B*M*U1]
  const int m  = (r0 / U1N) & 3;

  // phase 1: mid tiles (2 column tiles per wave)
#pragma unroll
  for (int t = 0; t < 2; ++t) {
    const int n0 = (wid * 2 + t) * 16;
    const int n  = n0 + lo;
    v8f acc = {};
    const float* Arow = fused + (size_t)(r0 + lo) * FF;
    const float* Brow = W_mid + (size_t)n * FF;
#pragma unroll
    for (int k0 = 0; k0 < FF; k0 += 32)
      acc = wmma_bf16(afrag_f32(Arow + k0 + klo), bfrag_f32(Brow + k0 + kbB), acc);
    const float bm_ = b_mid[n];
#pragma unroll
    for (int g = 0; g < 8; ++g) {
      const float x = acc[g] + bm_;
      const float ge = 0.5f * x * (1.0f + erff(x * 0.70710678118f));  // exact gelu
      s_mid[g + 8 * hi][n] = (__bf16)ge;
    }
  }
  __syncthreads();

  // phase 2: per-mode grouped output GEMM
#pragma unroll
  for (int t = 0; t < 2; ++t) {
    const int n0 = (wid * 2 + t) * 16;
    const int n  = n0 + lo;
    v8f acc = {};
    const float* Brow = W_out + ((size_t)m * FF + n) * FF;
#pragma unroll
    for (int k0 = 0; k0 < FF; k0 += 32)
      acc = wmma_bf16(afrag_bf16(&s_mid[lo][k0 + klo]), bfrag_f32(Brow + k0 + kbB), acc);
    const float bo = b_out[m * FF + n];
#pragma unroll
    for (int g = 0; g < 8; ++g) s_out[g + 8 * hi][n] = acc[g] + bo;
  }
  __syncthreads();

  // phase 3: layernorm over F=256 per row (16 threads per row)
  const int r = threadIdx.x >> 4;
  const int p = threadIdx.x & 15;
  float s1 = 0.0f, s2 = 0.0f;
#pragma unroll
  for (int c = 0; c < 16; ++c) {
    const float x = s_out[r][p * 16 + c];
    s1 += x; s2 += x * x;
  }
  s_r1[r][p] = s1; s_r2[r][p] = s2;
  __syncthreads();
  float mu = 0.0f, qq = 0.0f;
#pragma unroll
  for (int w = 0; w < 16; ++w) { mu += s_r1[r][w]; qq += s_r2[r][w]; }
  mu *= (1.0f / 256.0f);
  const float var = qq * (1.0f / 256.0f) - mu * mu;
  const float rstd = rsqrtf(var + 1e-12f);
  float* orow = outn + (size_t)(r0 + r) * FF;
#pragma unroll
  for (int c = 0; c < 16; ++c) {
    const int col = p * 16 + c;
    orow[col] = (s_out[r][col] - mu) * rstd * ln_g[col] + ln_b[col];
  }
}

// ---------------------------------------------------------------------------
// Kernel 4: learned soft-aggregate over modes. One wave per (b, i).
// ---------------------------------------------------------------------------
__global__ __launch_bounds__(128) void cafe_agg(
    const float* __restrict__ outn, const float* __restrict__ W_score,
    const float* __restrict__ b_score, float* __restrict__ out) {
  const int lane = threadIdx.x & 31;
  const int gw = blockIdx.x * 4 + (threadIdx.x >> 5);
  const int b = gw / U1N;
  const int i = gw % U1N;

  float sc[4];
#pragma unroll
  for (int m = 0; m < 4; ++m) {
    const float* row = outn + ((size_t)(b * MM + m) * U1N + i) * FF;
    float part = 0.0f;
#pragma unroll
    for (int t = 0; t < 8; ++t) part += row[lane * 8 + t] * W_score[lane * 8 + t];
    part += __shfl_xor(part, 1, 32);
    part += __shfl_xor(part, 2, 32);
    part += __shfl_xor(part, 4, 32);
    part += __shfl_xor(part, 8, 32);
    part += __shfl_xor(part, 16, 32);
    sc[m] = part + b_score[0];
  }
  const float mx = fmaxf(fmaxf(sc[0], sc[1]), fmaxf(sc[2], sc[3]));
  float pw[4], sum = 0.0f;
#pragma unroll
  for (int m = 0; m < 4; ++m) { pw[m] = __expf(sc[m] - mx); sum += pw[m]; }
  const float rinv = 1.0f / sum;
#pragma unroll
  for (int t = 0; t < 8; ++t) {
    const int f = lane * 8 + t;
    float v = 0.0f;
#pragma unroll
    for (int m = 0; m < 4; ++m)
      v += outn[((size_t)(b * MM + m) * U1N + i) * FF + f] * pw[m];
    out[((size_t)b * U1N + i) * FF + f] = v * rinv;
  }
}

// ---------------------------------------------------------------------------
extern "C" void kernel_launch(void* const* d_in, const int* in_sizes, int n_in,
                              void* d_out, int out_size, void* d_ws, size_t ws_size,
                              hipStream_t stream) {
  (void)in_sizes; (void)n_in; (void)out_size; (void)ws_size;
  const float* query   = (const float*)d_in[0];
  const float* key     = (const float*)d_in[1];
  const float* Wq      = (const float*)d_in[2];
  const float* bq      = (const float*)d_in[3];
  const float* Wk      = (const float*)d_in[4];
  const float* bk      = (const float*)d_in[5];
  const float* Wv      = (const float*)d_in[6];
  const float* W_mid   = (const float*)d_in[7];
  const float* b_mid   = (const float*)d_in[8];
  const float* W_out   = (const float*)d_in[9];
  const float* b_out   = (const float*)d_in[10];
  const float* ln_g    = (const float*)d_in[11];
  const float* ln_b    = (const float*)d_in[12];
  const float* W_score = (const float*)d_in[13];
  const float* b_score = (const float*)d_in[14];
  float* out = (float*)d_out;

  char* ws = (char*)d_ws;
  size_t off = 0;
  auto carve = [&](size_t bytes) -> char* {
    char* p = ws + off;
    off += (bytes + 255) & ~(size_t)255;
    return p;
  };
  __bf16* qb    = (__bf16*)carve((size_t)BB * U1N * CC * 2);           //  4 MB
  __bf16* kbuf  = (__bf16*)carve((size_t)BB * U2N * CC * 2);           //  4 MB
  __bf16* vt    = (__bf16*)carve((size_t)BB * MM * FF * U2N * 2);      // 16 MB
  float*  fused = (float*) carve((size_t)BB * MM * U1N * FF * 4);      // 32 MB
  float*  outn  = (float*) carve((size_t)BB * MM * U1N * FF * 4);      // 32 MB

  // Q, K projections (tied weights), V projection stored transposed.
  cafe_proj_gemm<<<2048, 128, 0, stream>>>(query, Wq, bq, qb,   BB * U1N, 256, 0);
  cafe_proj_gemm<<<2048, 128, 0, stream>>>(key,   Wk, bk, kbuf, BB * U2N, 256, 0);
  cafe_proj_gemm<<<8192, 128, 0, stream>>>(key,   Wv, nullptr, vt, BB * U2N, 1024, 1);

  // Attention: B*M*(U1/16) = 2048 workgroups of 8 waves.
  cafe_attn<<<2048, 256, 0, stream>>>(qb, kbuf, vt, fused);

  // Mid + per-mode output + layernorm: 1024 workgroups x 16 rows = (B*M*U1)/2 rows.
  cafe_mid_out<<<1024, 256, 0, stream>>>(fused, W_mid, b_mid, W_out, b_out,
                                         ln_g, ln_b, outn);

  // Mode soft-aggregate.
  cafe_agg<<<2048, 128, 0, stream>>>(outn, W_score, b_score, out);
}
